// MGCN_31507880083475
// MI455X (gfx1250) — matrix operand order repaired and checked by reading, not compiled, not run on hardware
//
#include <hip/hip_runtime.h>
#include <hip/hip_bf16.h>

// ---------------------------------------------------------------------------
// MGCN on MI455X (gfx1250): fused gather -> bf16 WMMA GEMM -> atomic scatter.
//
// Layer: out[i] = (1/deg[i]) * ( x[i] @ W[16]  +  sum_{edges j->i} x[j] @ W[et] )
// Implemented as:
//   1) dense GEMM (self loop) initializes out            (plain f32 stores)
//   2) per-relation fused edge tiles atomically add      (global_atomic_add_f32,
//      out is 51 MB -> resident in 192 MB L2)
//   3) scale by 1/deg, emit next-layer bf16 x (or final f32 output)
// Matmuls use v_wmma_f32_16x16x32_bf16 (f32 accumulate).
// ---------------------------------------------------------------------------

typedef unsigned short ushort_t;
typedef __attribute__((ext_vector_type(16))) __bf16 v16bf;
typedef __attribute__((ext_vector_type(8)))  float  v8f;

union FragU { uint4 q[2]; v16bf v; };

#define NSUB   50000
#define ESUB   400000
#define NREL   17          // 16 relations + self-loop matrix
#define DIM    256
#define CAP    (ESUB + NREL * 32)   // 400544, bucket-padded edge capacity
#define TILES  (CAP / 32)           // 12517 tiles of 32 edges
#define TRASH  NSUB                 // dump row for padding edges

// meta layout (ints): [0..16] cntRel, [17..34] padded offsets, [35..51] cursors
#define M_CNT  0
#define M_OFF  17
#define M_CUR  35

__device__ __forceinline__ ushort_t f2bf(float f) {
  unsigned u = __float_as_uint(f);
  unsigned r = u + 0x7FFFu + ((u >> 16) & 1u);   // round-to-nearest-even
  return (ushort_t)(r >> 16);
}

// ------------------------------ preprocessing ------------------------------

__global__ void init_meta_k(int* meta) {
  int t = threadIdx.x;
  if (t < 52) meta[t] = 0;
}

__global__ void init_deg_k(int* deg) {
  int n = blockIdx.x * 256 + threadIdx.x;
  if (n < NSUB) deg[n] = 1;                      // self loop
}

__global__ void prep_count_k(const int* __restrict__ edge_attr,
                             const int* __restrict__ e_id,
                             const int* __restrict__ edge_index,
                             int* meta, int* deg) {
  int e = blockIdx.x * 256 + threadIdx.x;
  if (e >= ESUB) return;
  int et = edge_attr[2 * e_id[e] + 1];
  atomicAdd(&meta[M_CNT + et], 1);
  atomicAdd(&deg[edge_index[ESUB + e]], 1);      // dst row
}

__global__ void scan_k(int* meta) {
  if (threadIdx.x == 0 && blockIdx.x == 0) {
    int off = 0;
    meta[M_OFF + 0] = 0;
    for (int r = 0; r < 16; ++r) {               // only 16 real relations carry edges
      off += ((meta[M_CNT + r] + 31) >> 5) << 5; // pad to 32-edge tiles
      meta[M_OFF + r + 1] = off;
      meta[M_CUR + r] = 0;
    }
    meta[M_OFF + 17] = off;                      // beyond: trash region (r treated as 16)
    meta[M_OFF + 16 + 1] = off;
  }
}

__global__ void init_buckets_k(int* bsrc, int* bdst) {
  int i = blockIdx.x * 256 + threadIdx.x;
  if (i < CAP) { bsrc[i] = 0; bdst[i] = TRASH; }
}

__global__ void scatter_edges_k(const int* __restrict__ edge_attr,
                                const int* __restrict__ e_id,
                                const int* __restrict__ edge_index,
                                int* meta, int* bsrc, int* bdst) {
  int e = blockIdx.x * 256 + threadIdx.x;
  if (e >= ESUB) return;
  int et = edge_attr[2 * e_id[e] + 1];
  int p = atomicAdd(&meta[M_CUR + et], 1);
  int i = meta[M_OFF + et] + p;
  bsrc[i] = edge_index[e];
  bdst[i] = edge_index[ESUB + e];
}

__global__ void inv_deg_k(const int* deg, float* inv) {
  int n = blockIdx.x * 256 + threadIdx.x;
  if (n < NSUB) inv[n] = 1.0f / (float)deg[n];
}

__global__ void gather_x_k(const float* __restrict__ entity,
                           const int* __restrict__ n_id,
                           ushort_t* __restrict__ xbf) {
  int i = blockIdx.x * 256 + threadIdx.x;
  if (i >= NSUB * DIM) return;
  int n = i >> 8, k = i & 255;
  xbf[i] = f2bf(entity[(size_t)n_id[n] * DIM + k]);
}

// rel matrix [r][k][n] f32  ->  packed bf16 [r][n][k] (k contiguous for WMMA B)
__global__ void pack_w_k(const float* __restrict__ rm, ushort_t* __restrict__ wp) {
  int i = blockIdx.x * 256 + threadIdx.x;
  if (i >= NREL * DIM * DIM) return;
  int r = i >> 16, rem = i & 65535, n = rem >> 8, k = rem & 255;
  wp[((size_t)r << 16) + n * DIM + k] = f2bf(rm[((size_t)r << 16) + k * DIM + n]);
}

// ------------------------- WMMA fragment loaders ---------------------------
// A (16x32 bf16): lane(0-15)=row M, half=lane>>4:
//   elems 0..7  <- K = k0 + half*8 .. +8      (contiguous)
//   elems 8..15 <- K = k0 + 16 + half*8 .. +8 (contiguous)
// B (32x16 bf16), LDS stored n-major [n][k]: lane&15 = column n, half=lane>>4:
//   elems 0..7  <- K = k0 + half*8 .. +8      (contiguous)
//   elems 8..15 <- K = k0 + 16 + half*8 .. +8 (contiguous)
__device__ __forceinline__ v16bf load_a(const ushort_t* rowp, int k0, int half) {
  FragU f;
  f.q[0] = *(const uint4*)(rowp + k0 + half * 8);
  f.q[1] = *(const uint4*)(rowp + k0 + 16 + half * 8);
  return f.v;
}
__device__ __forceinline__ v16bf load_b(const ushort_t* colp, int k0, int half) {
  FragU f;
  f.q[0] = *(const uint4*)(colp + k0 + half * 8);
  f.q[1] = *(const uint4*)(colp + k0 + 16 + half * 8);
  return f.v;
}

// --------------------- dense self-loop GEMM: out = x @ W[16] ---------------
// grid (ceil(N/32), 4), block 256 (8 waves). Tile: 32 rows x 64 cols.
__global__ __launch_bounds__(256) void selfloop_gemm_k(
    const ushort_t* __restrict__ xbf, const ushort_t* __restrict__ w16,
    float* __restrict__ out) {
  __shared__ alignas(16) ushort_t sW[64 * DIM];   // 32 KB
  __shared__ alignas(16) ushort_t sA[32 * DIM];   // 16 KB
  int rt = blockIdx.x, cg = blockIdx.y, t = threadIdx.x;

  { // stage W[16] column chunk (64 n-rows, k contiguous)
    const uint4* s = (const uint4*)(w16 + (size_t)cg * 64 * DIM);
    uint4* d = (uint4*)sW;
    #pragma unroll
    for (int i = 0; i < 8; ++i) d[t + 256 * i] = s[t + 256 * i];
  }
  { // stage 32 contiguous x rows (clamped)
    int m = t >> 3, j = t & 7;
    int row = rt * 32 + m; if (row >= NSUB) row = NSUB - 1;
    const uint4* s = (const uint4*)(xbf + (size_t)row * DIM + j * 32);
    uint4* d = (uint4*)(sA + m * DIM + j * 32);
    d[0] = s[0]; d[1] = s[1]; d[2] = s[2]; d[3] = s[3];
  }
  __syncthreads();

  int wave = t >> 5, lane = t & 31;
  int rowSub = wave & 1, colSub = wave >> 1;
  int half = lane >> 4, l15 = lane & 15;
  const ushort_t* ap = sA + (rowSub * 16 + l15) * DIM;
  const ushort_t* bp = sW + (colSub * 16 + l15) * DIM;
  v8f c = {};
  #pragma unroll
  for (int kb = 0; kb < 8; ++kb) {
    v16bf a = load_a(ap, kb * 32, half);
    v16bf b = load_b(bp, kb * 32, half);
    c = __builtin_amdgcn_wmma_f32_16x16x32_bf16(false, a, false, b, (short)0, c,
                                                false, false);
  }
  int col = cg * 64 + colSub * 16 + l15;
  #pragma unroll
  for (int v = 0; v < 8; ++v) {
    int gr = rt * 32 + rowSub * 16 + half * 8 + v;   // C: VGPR v -> M = v + half*8
    if (gr < NSUB) out[(size_t)gr * DIM + col] = c[v];
  }
}

// ---------------- fused per-edge-tile GEMM + atomic scatter ----------------
// grid (TILES, 4), block 256. Tile: 32 edges x 64 cols; W[r] chunk in LDS.
__global__ __launch_bounds__(256) void edge_gemm_k(
    const ushort_t* __restrict__ xbf, const ushort_t* __restrict__ wp,
    const int* __restrict__ bsrc, const int* __restrict__ bdst,
    const int* __restrict__ offPad, float* __restrict__ out) {
  __shared__ alignas(16) ushort_t sW[64 * DIM];
  __shared__ alignas(16) ushort_t sA[32 * DIM];
  __shared__ int sdst[32];
  int tile = blockIdx.x, cg = blockIdx.y, t = threadIdx.x;
  int e0 = tile * 32;

  // which relation owns this tile (buckets padded to 32-edge multiples;
  // trailing trash region resolves to r=16 with dst=TRASH -> harmless)
  int r = 16;
  for (int i = 0; i < 16; ++i)
    if (e0 >= offPad[i] && e0 < offPad[i + 1]) { r = i; break; }

  { // stage W[r] column chunk
    const uint4* s = (const uint4*)(wp + ((size_t)r << 16) + (size_t)cg * 64 * DIM);
    uint4* d = (uint4*)sW;
    #pragma unroll
    for (int i = 0; i < 8; ++i) d[t + 256 * i] = s[t + 256 * i];
  }
  if (t < 32) sdst[t] = bdst[e0 + t];
  { // gather 32 source rows
    int m = t >> 3, j = t & 7;
    int srow = bsrc[e0 + m];
    const uint4* s = (const uint4*)(xbf + (size_t)srow * DIM + j * 32);
    uint4* d = (uint4*)(sA + m * DIM + j * 32);
    d[0] = s[0]; d[1] = s[1]; d[2] = s[2]; d[3] = s[3];
  }
  __syncthreads();

  int wave = t >> 5, lane = t & 31;
  int rowSub = wave & 1, colSub = wave >> 1;
  int half = lane >> 4, l15 = lane & 15;
  const ushort_t* ap = sA + (rowSub * 16 + l15) * DIM;
  const ushort_t* bp = sW + (colSub * 16 + l15) * DIM;
  v8f c = {};
  #pragma unroll
  for (int kb = 0; kb < 8; ++kb) {
    v16bf a = load_a(ap, kb * 32, half);
    v16bf b = load_b(bp, kb * 32, half);
    c = __builtin_amdgcn_wmma_f32_16x16x32_bf16(false, a, false, b, (short)0, c,
                                                false, false);
  }
  int col = cg * 64 + colSub * 16 + l15;
  #pragma unroll
  for (int v = 0; v < 8; ++v) {
    int m = rowSub * 16 + half * 8 + v;
    int d = sdst[m];                                  // TRASH row for padding
    atomicAdd(&out[(size_t)d * DIM + col], c[v]);     // L2-resident accumulator
  }
}

// -------------------------- scale by 1/deg ---------------------------------
__global__ void scale_k(const float* __restrict__ out, const float* __restrict__ inv,
                        ushort_t* __restrict__ xbf, float* __restrict__ fout,
                        int final_layer) {
  int i = blockIdx.x * 256 + threadIdx.x;
  if (i >= NSUB * DIM) return;
  float v = out[i] * inv[i >> 8];
  if (final_layer) fout[i] = v;
  else             xbf[i]  = f2bf(v);
}

// ------------------------------- launcher ----------------------------------

extern "C" void kernel_launch(void* const* d_in, const int* in_sizes, int n_in,
                              void* d_out, int out_size, void* d_ws, size_t ws_size,
                              hipStream_t stream) {
  const int*   edge_attr  = (const int*)  d_in[0];
  const int*   e_id       = (const int*)  d_in[1];
  const int*   n_id       = (const int*)  d_in[2];
  const int*   edge_index = (const int*)  d_in[3];
  const float* entity     = (const float*)d_in[4];
  // d_in[5] relation_embedding: dead code in reference
  const float* rm1        = (const float*)d_in[6];
  const float* rm2        = (const float*)d_in[7];
  float* outp = (float*)d_out;

  char* ws = (char*)d_ws;
  size_t o = 0;
  auto take = [&](size_t bytes) { size_t a = o; o = (o + bytes + 255) & ~(size_t)255; return a; };
  ushort_t* xbf  = (ushort_t*)(ws + take((size_t)NSUB * DIM * 2));
  float* outacc  = (float*)   (ws + take((size_t)(NSUB + 1) * DIM * 4));
  ushort_t* wp1  = (ushort_t*)(ws + take((size_t)NREL * DIM * DIM * 2));
  ushort_t* wp2  = (ushort_t*)(ws + take((size_t)NREL * DIM * DIM * 2));
  int* bsrc      = (int*)     (ws + take((size_t)CAP * 4));
  int* bdst      = (int*)     (ws + take((size_t)CAP * 4));
  int* deg       = (int*)     (ws + take((size_t)NSUB * 4));
  float* inv     = (float*)   (ws + take((size_t)NSUB * 4));
  int* meta      = (int*)     (ws + take(64 * 4));

  // ---- preprocess (all stream-ordered for graph capture) ----
  init_meta_k<<<1, 64, 0, stream>>>(meta);
  init_deg_k<<<(NSUB + 255) / 256, 256, 0, stream>>>(deg);
  prep_count_k<<<(ESUB + 255) / 256, 256, 0, stream>>>(edge_attr, e_id, edge_index, meta, deg);
  scan_k<<<1, 32, 0, stream>>>(meta);
  init_buckets_k<<<(CAP + 255) / 256, 256, 0, stream>>>(bsrc, bdst);
  scatter_edges_k<<<(ESUB + 255) / 256, 256, 0, stream>>>(edge_attr, e_id, edge_index, meta, bsrc, bdst);
  inv_deg_k<<<(NSUB + 255) / 256, 256, 0, stream>>>(deg, inv);
  gather_x_k<<<(NSUB * DIM + 255) / 256, 256, 0, stream>>>(entity, n_id, xbf);
  pack_w_k<<<(NREL * DIM * DIM + 255) / 256, 256, 0, stream>>>(rm1, wp1);
  pack_w_k<<<(NREL * DIM * DIM + 255) / 256, 256, 0, stream>>>(rm2, wp2);

  // ---- 3 conv layers ----
  const int mtiles = (NSUB + 31) / 32;
  for (int l = 0; l < 3; ++l) {
    const ushort_t* wp = (l < 2) ? wp1 : wp2;
    selfloop_gemm_k<<<dim3(mtiles, 4), 256, 0, stream>>>(xbf, wp + ((size_t)16 << 16), outacc);
    edge_gemm_k<<<dim3(TILES, 4), 256, 0, stream>>>(xbf, wp, bsrc, bdst, meta + M_OFF, outacc);
    scale_k<<<(NSUB * DIM + 255) / 256, 256, 0, stream>>>(outacc, inv, xbf, outp, (l == 2) ? 1 : 0);
  }
}
